// JTNNEncoder_46720654246403
// MI455X (gfx1250) — matrix operand order repaired and checked by reading, not compiled, not run on hardware
//
#include <hip/hip_runtime.h>

#define HDIM 256
#define INDIM 288

typedef __attribute__((ext_vector_type(16))) __bf16 v16bf;
typedef __attribute__((ext_vector_type(8)))  float  v8f;

__device__ __forceinline__ unsigned short f2bf(float x){
    unsigned int u = __float_as_uint(x);
    return (unsigned short)((u + 0x7FFFu + ((u >> 16) & 1u)) >> 16);  // RNE
}
__device__ __forceinline__ float bf2f(unsigned short b){
    return __uint_as_float(((unsigned int)b) << 16);
}
__device__ __forceinline__ float sigmoidf_(float x){
    return 1.0f / (1.0f + __expf(-x));
}

// Load a bf16 WMMA operand fragment: 16 halves per lane = two 16-byte chunks.
__device__ __forceinline__ v16bf ld_frag(const unsigned short* p0, const unsigned short* p1){
    union { v16bf v; uint4 q[2]; } u;
    u.q[0] = *(const uint4*)p0;
    u.q[1] = *(const uint4*)p1;
    return u.v;
}
__device__ __forceinline__ v8f wmma_bf16(v16bf a, v16bf b, v8f c){
    return __builtin_amdgcn_wmma_f32_16x16x32_bf16(false, a, false, b, (short)0, c, false, false);
}

// 16-row A (row-major bf16, stride lda) x W^T (W row-major [256,ldw] bf16) -> 4 16x16 tiles per wave.
// A layout: lane&15 = row; half=lane>>4: K = {k0+half*8 .. +7} and {k0+16+half*8 .. +7}.
// B layout: lane&15 = out col; K = {k0+half*8 .. +7} and {k0+16+half*8 .. +7} (same as A).
__device__ __forceinline__ void gemm_tile(const unsigned short* A, int lda,
                                          const unsigned short* W, int ldw, int nk,
                                          int wv, int l15, int half, v8f* acc){
    const unsigned short* arow = A + (size_t)l15 * lda;
    const unsigned short* w0  = W + (size_t)(wv * 64 + l15) * ldw;
    for (int kk = 0; kk < nk; ++kk){
        int k0 = kk * 32;
        v16bf a = ld_frag(arow + k0 + half*8, arow + k0 + 16 + half*8);
#pragma unroll
        for (int t = 0; t < 4; ++t){
            const unsigned short* wr = w0 + (size_t)t * 16 * ldw;
            v16bf b = ld_frag(wr + k0 + half*8, wr + k0 + 16 + half*8);
            acc[t] = wmma_bf16(a, b, acc[t]);
        }
    }
}

// Async DMA gather of a 16-row x 256-col bf16 tile (rows indexed by nbk[]) into LDS.
// Each of 128 threads issues 4 GLOBAL_LOAD_ASYNC_TO_LDS_B128 (tracked by ASYNCcnt).
// LDS byte address = low 32 bits of the generic pointer (LDS aperture rule).
__device__ __forceinline__ void gather_async(const unsigned short* __restrict__ hsrc,
                                             unsigned short* dst, const int* nbk, int tid){
#pragma unroll
    for (int c = 0; c < 4; ++c){
        int ch = tid + c * 128;            // 0..511 chunks of 16B
        int r  = ch >> 5, cc = ch & 31;
        const unsigned short* g = hsrc + (size_t)nbk[r] * HDIM + cc * 8;
        unsigned lds = (unsigned)(size_t)(dst + r * 264 + cc * 8);
        asm volatile("global_load_async_to_lds_b128 %0, %1, off"
                     :: "v"(lds), "v"(g) : "memory");
    }
}

// ---------------- setup kernels ----------------
__global__ void zero_us(unsigned short* __restrict__ p, int n){
    for (int i = blockIdx.x*blockDim.x + threadIdx.x; i < n; i += gridDim.x*blockDim.x) p[i] = 0;
}

__global__ void build_fe(const int* __restrict__ fnode, const float* __restrict__ emb,
                         const float* __restrict__ c1, const float* __restrict__ c2,
                         unsigned short* __restrict__ fe, int n){
    int total = n * INDIM;
    for (int i = blockIdx.x*blockDim.x + threadIdx.x; i < total; i += gridDim.x*blockDim.x){
        int r = i / INDIM, c = i - r * INDIM;
        float v;
        if (c < 256)      v = emb[(size_t)fnode[r]*256 + c];
        else if (c < 272) v = c1[r*16 + (c - 256)];
        else              v = c2[r*16 + (c - 272)];
        fe[i] = f2bf(v);
    }
}

__global__ void cvt_w(const float* __restrict__ src, unsigned short* __restrict__ dst,
                      int rows, int cols, int ld, int col0){
    int total = rows * cols;
    for (int i = blockIdx.x*blockDim.x + threadIdx.x; i < total; i += gridDim.x*blockDim.x){
        int r = i / cols, c = i - r * cols;
        dst[i] = f2bf(src[(size_t)r*ld + col0 + c]);
    }
}

// ---------------- loop-invariant precompute: xWr / xWz / xWh ----------------
__global__ __launch_bounds__(128)
void xw_kernel(const unsigned short* __restrict__ fe, const int* __restrict__ fmess,
               const unsigned short* __restrict__ Wr_bf, const unsigned short* __restrict__ Wzx_bf,
               const unsigned short* __restrict__ Whx_bf,
               float* __restrict__ xWr, float* __restrict__ xWz, float* __restrict__ xWh){
    const int tid = threadIdx.x, lane = tid & 31, wv = tid >> 5, l15 = lane & 15, half = lane >> 4;
    const int m0 = blockIdx.x * 16;
    const unsigned short* W; float* out;
    if (blockIdx.y == 0){ W = Wr_bf;  out = xWr; }
    else if (blockIdx.y == 1){ W = Wzx_bf; out = xWz; }
    else { W = Whx_bf; out = xWh; }

    const unsigned short* arow = fe + (size_t)fmess[m0 + l15] * INDIM;   // gathered A row
    const unsigned short* w0 = W + (size_t)(wv * 64 + l15) * INDIM;
    v8f acc[4] = {};
    for (int kk = 0; kk < 9; ++kk){                                       // K = 288
        int k0 = kk * 32;
        v16bf a = ld_frag(arow + k0 + half*8, arow + k0 + 16 + half*8);
#pragma unroll
        for (int t = 0; t < 4; ++t){
            const unsigned short* wr = w0 + (size_t)t * 16 * INDIM;
            v16bf b = ld_frag(wr + k0 + half*8, wr + k0 + 16 + half*8);
            acc[t] = wmma_bf16(a, b, acc[t]);
        }
    }
#pragma unroll
    for (int t = 0; t < 4; ++t){
        int col = wv * 64 + t * 16 + l15;
#pragma unroll
        for (int v = 0; v < 8; ++v)
            out[(size_t)(m0 + half*8 + v) * HDIM + col] = acc[t][v];
    }
}

// ---------------- one GRU message-passing step (h stored bf16 between steps) ----------------
__global__ __launch_bounds__(128)
void step_kernel(const unsigned short* __restrict__ h_in, unsigned short* __restrict__ h_out,
                 float* __restrict__ h_out_f32,                      // non-null only on final step
                 const unsigned short* __restrict__ Ur_bf, const unsigned short* __restrict__ Wzh_bf,
                 const unsigned short* __restrict__ Whh_bf,
                 const float* __restrict__ xWr, const float* __restrict__ xWz, const float* __restrict__ xWh,
                 const float* __restrict__ Ur_b, const float* __restrict__ Wz_b, const float* __restrict__ Wh_b,
                 const int* __restrict__ mess_graph){
    __shared__ alignas(16) unsigned short hbf[2][16][264];   // double-buffered async gather tiles
    __shared__ alignas(16) unsigned short stage[16][264];    // sum_h / sum_gated staging for WMMA
    __shared__ int nb[5][16];

    const int tid = threadIdx.x, lane = tid & 31, wv = tid >> 5, l15 = lane & 15, half = lane >> 4;
    const int m0 = blockIdx.x * 16;

    if (tid < 80) nb[tid % 5][tid / 5] = mess_graph[(m0 + tid / 5) * 5 + (tid % 5)];
    __syncthreads();

    gather_async(h_in, &hbf[0][0][0], &nb[0][0], tid);       // prefetch neighbor 0

    float sumh[4][8], sumg[4][8];
#pragma unroll
    for (int t = 0; t < 4; ++t)
#pragma unroll
        for (int v = 0; v < 8; ++v){ sumh[t][v] = 0.0f; sumg[t][v] = 0.0f; }

    for (int k = 0; k < 5; ++k){
        if (k < 4){
            gather_async(h_in, &hbf[(k + 1) & 1][0][0], &nb[k + 1][0], tid);
            asm volatile("s_wait_asynccnt 0x4" ::: "memory");   // neighbor k done; k+1 in flight
        } else {
            asm volatile("s_wait_asynccnt 0x0" ::: "memory");
        }
        __syncthreads();

        const unsigned short* A = &hbf[k & 1][0][0];
        v8f acc[4] = {};
        gemm_tile(A, 264, Ur_bf, 256, 8, wv, l15, half, acc);   // h_nei @ Ur^T
#pragma unroll
        for (int t = 0; t < 4; ++t){
            int col = wv * 64 + t * 16 + l15;
            float bias = Ur_b[col];
#pragma unroll
            for (int v = 0; v < 8; ++v){
                int rl = half*8 + v;
                float hv = bf2f(A[rl * 264 + col]);
                sumh[t][v] += hv;
                float r = sigmoidf_(acc[t][v] + xWr[(size_t)(m0 + rl) * HDIM + col] + bias);
                sumg[t][v] += r * hv;
            }
        }
        __syncthreads();   // buffer (k&1) free for gather of neighbor k+2
    }

    // z = sigmoid(xWz + sum_h @ Wzh^T + bz)
#pragma unroll
    for (int t = 0; t < 4; ++t)
#pragma unroll
        for (int v = 0; v < 8; ++v)
            stage[half*8 + v][wv*64 + t*16 + l15] = f2bf(sumh[t][v]);
    __syncthreads();
    v8f zacc[4] = {};
    gemm_tile(&stage[0][0], 264, Wzh_bf, 256, 8, wv, l15, half, zacc);
    float zreg[4][8];
#pragma unroll
    for (int t = 0; t < 4; ++t){
        int col = wv * 64 + t * 16 + l15;
        float bz = Wz_b[col];
#pragma unroll
        for (int v = 0; v < 8; ++v)
            zreg[t][v] = sigmoidf_(zacc[t][v] + xWz[(size_t)(m0 + half*8 + v) * HDIM + col] + bz);
    }
    __syncthreads();

    // pre_h = tanh(xWh + sum_gated @ Whh^T + bh);  h_new = (1-z)*sum_h + z*pre_h
#pragma unroll
    for (int t = 0; t < 4; ++t)
#pragma unroll
        for (int v = 0; v < 8; ++v)
            stage[half*8 + v][wv*64 + t*16 + l15] = f2bf(sumg[t][v]);
    __syncthreads();
    v8f pacc[4] = {};
    gemm_tile(&stage[0][0], 264, Whh_bf, 256, 8, wv, l15, half, pacc);
#pragma unroll
    for (int t = 0; t < 4; ++t){
        int col = wv * 64 + t * 16 + l15;
        float bh = Wh_b[col];
#pragma unroll
        for (int v = 0; v < 8; ++v){
            int rl = half*8 + v, m = m0 + rl;
            float pre = tanhf(pacc[t][v] + xWh[(size_t)m * HDIM + col] + bh);
            float z = zreg[t][v];
            float hn = (1.0f - z) * sumh[t][v] + z * pre;
            if (m == 0) hn = 0.0f;                            // padding message stays zero
            h_out[(size_t)m * HDIM + col] = f2bf(hn);
            if (h_out_f32) h_out_f32[(size_t)m * HDIM + col] = hn;
        }
    }
}

// ---------------- output layer: only the 64 scope_start rows ----------------
__global__ __launch_bounds__(128)
void out_kernel(const unsigned short* __restrict__ fe, const float* __restrict__ hfin,
                const int* __restrict__ node_graph, const int* __restrict__ scope,
                const unsigned short* __restrict__ Wout_bf, const float* __restrict__ out_b,
                float* __restrict__ tree){
    __shared__ alignas(16) unsigned short Abf[16][552];   // [fnode_e | mess_nei] bf16, K = 544
    __shared__ int sidx[16];
    const int tid = threadIdx.x, lane = tid & 31, wv = tid >> 5, l15 = lane & 15, half = lane >> 4;
    const int m0 = blockIdx.x * 16;
    if (tid < 16) sidx[tid] = scope[m0 + tid];
    __syncthreads();
    for (int i = tid; i < 16*INDIM; i += 128){
        int r = i / INDIM, c = i - r * INDIM;
        Abf[r][c] = fe[(size_t)sidx[r] * INDIM + c];
    }
    for (int i = tid; i < 16*256; i += 128){
        int r = i >> 8, c = i & 255;
        const int* ng = node_graph + (size_t)sidx[r] * 6;
        float s = 0.0f;
#pragma unroll
        for (int j = 0; j < 6; ++j) s += hfin[(size_t)ng[j] * HDIM + c];
        Abf[r][INDIM + c] = f2bf(s);
    }
    __syncthreads();
    v8f acc[4] = {};
    gemm_tile(&Abf[0][0], 552, Wout_bf, 544, 17, wv, l15, half, acc);    // K = 544
#pragma unroll
    for (int t = 0; t < 4; ++t){
        int col = wv * 64 + t * 16 + l15;
        float b = out_b[col];
#pragma unroll
        for (int v = 0; v < 8; ++v){
            float val = acc[t][v] + b;
            tree[(size_t)(m0 + half*8 + v) * HDIM + col] = val > 0.0f ? val : 0.0f;
        }
    }
}

extern "C" void kernel_launch(void* const* d_in, const int* in_sizes, int n_in,
                              void* d_out, int out_size, void* d_ws, size_t ws_size,
                              hipStream_t stream){
    const int*   fnode      = (const int*)d_in[0];
    const int*   fmess      = (const int*)d_in[1];
    const int*   node_graph = (const int*)d_in[2];
    const int*   mess_graph = (const int*)d_in[3];
    const int*   scope      = (const int*)d_in[4];
    const float* c1    = (const float*)d_in[5];
    const float* c2    = (const float*)d_in[6];
    const float* emb   = (const float*)d_in[7];
    const float* Wz_w  = (const float*)d_in[8];
    const float* Wz_b  = (const float*)d_in[9];
    const float* Wr_w  = (const float*)d_in[10];
    const float* Ur_w  = (const float*)d_in[11];
    const float* Ur_b  = (const float*)d_in[12];
    const float* Wh_w  = (const float*)d_in[13];
    const float* Wh_b  = (const float*)d_in[14];
    const float* out_w = (const float*)d_in[15];
    const float* out_b = (const float*)d_in[16];

    const int n = in_sizes[0];   // 30000 nodes
    const int m = in_sizes[1];   // 40000 messages
    const int b = in_sizes[4];   // 64 trees

    float* tree  = (float*)d_out;
    float* hOutF = tree + (size_t)b * HDIM;      // fp32 h output (written by the final step only)

    char* ws = (char*)d_ws;
    size_t off = 0;
    unsigned short* fe  = (unsigned short*)(ws + off); off += (size_t)n * INDIM * 2;
    float* xWr = (float*)(ws + off); off += (size_t)m * HDIM * 4;
    float* xWz = (float*)(ws + off); off += (size_t)m * HDIM * 4;
    float* xWh = (float*)(ws + off); off += (size_t)m * HDIM * 4;
    unsigned short* hP = (unsigned short*)(ws + off); off += (size_t)m * HDIM * 2;  // bf16 h ping
    unsigned short* hQ = (unsigned short*)(ws + off); off += (size_t)m * HDIM * 2;  // bf16 h pong
    unsigned short* Wr_bf   = (unsigned short*)(ws + off); off += 256*288*2;
    unsigned short* Wzx_bf  = (unsigned short*)(ws + off); off += 256*288*2;
    unsigned short* Wzh_bf  = (unsigned short*)(ws + off); off += 256*256*2;
    unsigned short* Whx_bf  = (unsigned short*)(ws + off); off += 256*288*2;
    unsigned short* Whh_bf  = (unsigned short*)(ws + off); off += 256*256*2;
    unsigned short* Ur_bf   = (unsigned short*)(ws + off); off += 256*256*2;
    unsigned short* Wout_bf = (unsigned short*)(ws + off); off += 256*544*2;

    // h0 = 0 (re-zeroed every launch -> deterministic across graph replays)
    zero_us<<<2048, 256, 0, stream>>>(hP, m * HDIM);
    build_fe<<<4096, 256, 0, stream>>>(fnode, emb, c1, c2, fe, n);
    cvt_w<<<288, 256, 0, stream>>>(Wr_w,  Wr_bf,   256, 288, 288,   0);
    cvt_w<<<288, 256, 0, stream>>>(Wz_w,  Wzx_bf,  256, 288, 544,   0);
    cvt_w<<<256, 256, 0, stream>>>(Wz_w,  Wzh_bf,  256, 256, 544, 288);
    cvt_w<<<288, 256, 0, stream>>>(Wh_w,  Whx_bf,  256, 288, 544,   0);
    cvt_w<<<256, 256, 0, stream>>>(Wh_w,  Whh_bf,  256, 256, 544, 288);
    cvt_w<<<256, 256, 0, stream>>>(Ur_w,  Ur_bf,   256, 256, 256,   0);
    cvt_w<<<544, 256, 0, stream>>>(out_w, Wout_bf, 256, 544, 544,   0);

    dim3 g1(m / 16, 3);
    xw_kernel<<<g1, 128, 0, stream>>>(fe, fmess, Wr_bf, Wzx_bf, Whx_bf, xWr, xWz, xWh);

    // 6 GRU steps; bf16 ping-pong, fp32 h written to d_out on the last step
    const unsigned short* hin = hP; unsigned short* hout = hQ;
    for (int s = 0; s < 6; ++s){
        float* f32out = (s == 5) ? hOutF : (float*)nullptr;
        step_kernel<<<m / 16, 128, 0, stream>>>(hin, hout, f32out, Ur_bf, Wzh_bf, Whh_bf,
                                                xWr, xWz, xWh, Ur_b, Wz_b, Wh_b, mess_graph);
        const unsigned short* t = hout; hout = (unsigned short*)hin; hin = t;
    }

    out_kernel<<<b / 16, 128, 0, stream>>>(fe, hOutF, node_graph, scope, Wout_bf, out_b, tree);
}
